// ElmanRNN_2723009265890
// MI455X (gfx1250) — matrix operand order repaired and checked by reading, not compiled, not run on hardware
//
#include <hip/hip_runtime.h>
#include <hip/hip_bf16.h>

// ---------------------------------------------------------------------------
// Elman RNN on MI455X (gfx1250): bf16 WMMA with f32 accumulation.
//   B=64, S=512, I=512, H=1024, O=512
// Phase 1: xW = x @ W_h^T   (parallel GEMM, M=B*S=32768; 16x64 per wave,
//                            one f32->bf16 A-conversion amortized over 4 WMMAs)
// Phase 2: h = tanh(xW_t + h U^T + bias), 512 sequential step kernels,
//          32x32 per wave (2 A frags x 2 B frags -> 4 WMMAs / K-step)
// Phase 3: y = h @ W_y^T + b_wy + b_y, 32x32 per wave
// ---------------------------------------------------------------------------

typedef __bf16 bf16_t;
typedef __attribute__((ext_vector_type(16))) __bf16 v16bf;
typedef __attribute__((ext_vector_type(8)))  __bf16 v8bf;
typedef __attribute__((ext_vector_type(8)))  float  v8f;
typedef __attribute__((ext_vector_type(4)))  float  f32x4;

#define RNN_B 64
#define RNN_S 512
#define RNN_I 512
#define RNN_H 1024
#define RNN_O 512

// ---- WMMA wrapper ----------------------------------------------------------
__device__ __forceinline__ v8f wmma_bf16(v16bf a, v16bf b, v8f c) {
  // (neg_a, A, neg_b, B, c_mod, C, reuse_a, reuse_b)
  return __builtin_amdgcn_wmma_f32_16x16x32_bf16(false, a, false, b,
                                                 (short)0, c, false, false);
}

// ---- A fragment: 16x32 tile from row-major bf16 [*, lda] -------------------
// lane l<16 : row m0+l,    K = {k0..k0+7,  k0+16..k0+23}
// lane l>=16: row m0+l-16, K = {k0+8..15,  k0+24..k0+31}
__device__ __forceinline__ v16bf load_a_bf16(const bf16_t* __restrict__ A,
                                             int lda, int row0, int k0, int lane) {
  const int r  = row0 + (lane & 15);
  const int kb = k0 + ((lane >> 4) << 3);
  union { v16bf v; v8bf h[2]; } u;
  u.h[0] = *(const v8bf*)(A + r * lda + kb);
  u.h[1] = *(const v8bf*)(A + r * lda + kb + 16);
  return u.v;
}

// Same tile from an f32 source (convert to bf16 on load).
__device__ __forceinline__ v16bf load_a_f32(const float* __restrict__ A,
                                            int lda, int row0, int k0, int lane) {
  const int r  = row0 + (lane & 15);
  const int kb = k0 + ((lane >> 4) << 3);
  const float* p = A + r * lda + kb;
  f32x4 a0 = *(const f32x4*)(p);
  f32x4 a1 = *(const f32x4*)(p + 4);
  f32x4 a2 = *(const f32x4*)(p + 16);
  f32x4 a3 = *(const f32x4*)(p + 20);
  v16bf v;
#pragma unroll
  for (int i = 0; i < 4; ++i) {
    v[i]      = (bf16_t)a0[i];
    v[i + 4]  = (bf16_t)a1[i];
    v[i + 8]  = (bf16_t)a2[i];
    v[i + 12] = (bf16_t)a3[i];
  }
  return v;
}

// ---- B fragment: 32x16 tile, B[k][n] = W[n][k], W row-major [N][K] bf16 ----
// lane l<16 : col n0+l,    K = {k0..k0+7,  k0+16..k0+23}
// lane l>=16: col n0+l-16, K = {k0+8..15,  k0+24..k0+31}
__device__ __forceinline__ v16bf load_b_w(const bf16_t* __restrict__ W,
                                          int ldw, int n0, int k0, int lane) {
  const int n  = n0 + (lane & 15);
  const int kb = k0 + ((lane >> 4) << 3);
  union { v16bf v; v8bf h[2]; } u;
  u.h[0] = *(const v8bf*)(W + n * ldw + kb);
  u.h[1] = *(const v8bf*)(W + n * ldw + kb + 16);
  return u.v;
}

__device__ __forceinline__ float fast_tanh(float v) {
  // tanh(v) = 1 - 2/(e^{2v}+1); saturates correctly as e -> 0 or +inf.
  float e = __expf(2.0f * v);
  return 1.0f - 2.0f / (e + 1.0f);
}

// ---- prep: convert weights f32->bf16 into ws, zero h0 ----------------------
__global__ void k_prep(const float* __restrict__ Wh, const float* __restrict__ Uh,
                       const float* __restrict__ Wy,
                       bf16_t* __restrict__ Whb, bf16_t* __restrict__ Uhb,
                       bf16_t* __restrict__ Wyb, bf16_t* __restrict__ h0) {
  const int nWh = RNN_H * RNN_I;
  const int nUh = RNN_H * RNN_H;
  const int nWy = RNN_O * RNN_H;
  const int nH  = RNN_B * RNN_H;
  int i = blockIdx.x * blockDim.x + threadIdx.x;
  if (i < nWh)                         Whb[i] = (bf16_t)Wh[i];
  else if (i < nWh + nUh)              Uhb[i - nWh] = (bf16_t)Uh[i - nWh];
  else if (i < nWh + nUh + nWy)        Wyb[i - nWh - nUh] = (bf16_t)Wy[i - nWh - nUh];
  else if (i < nWh + nUh + nWy + nH)   h0[i - nWh - nUh - nWy] = (bf16_t)0.0f;
}

// ---- phase 1: xW[s][b][h] = sum_i x[b][s][i] * Wh[h][i] --------------------
// 16x64 strip per wave: 1 A fragment (f32->bf16 once) feeds 4 WMMAs.
__global__ void __launch_bounds__(256)
k_xw(const float* __restrict__ x,       // [B*S, I] rows r = b*S + s
     const bf16_t* __restrict__ Whb,    // [H, I] bf16
     bf16_t* __restrict__ xw) {         // [S, B, H] bf16
  const int lane = threadIdx.x & 31;
  const int wave = threadIdx.x >> 5;
  const int w  = blockIdx.x * 8 + wave;            // 32768 wave-strips
  const int ng = w & 15;                           // H/64 = 16 N-groups
  const int mt = w >> 4;                           // (B*S)/16 = 2048 M-tiles
  const int m0 = mt * 16, n0 = ng * 64;

  v8f acc[4];
#pragma unroll
  for (int ni = 0; ni < 4; ++ni)
#pragma unroll
    for (int j = 0; j < 8; ++j) acc[ni][j] = 0.0f;

  for (int k0 = 0; k0 < RNN_I; k0 += 32) {
    v16bf a = load_a_f32(x, RNN_I, m0, k0, lane);  // converted once
#pragma unroll
    for (int ni = 0; ni < 4; ++ni) {
      v16bf b = load_b_w(Whb, RNN_I, n0 + ni * 16, k0, lane);
      acc[ni] = wmma_bf16(a, b, acc[ni]);
    }
  }

  const int rbase = m0 + ((lane >> 4) << 3);       // + j = global row (b*S+s)
#pragma unroll
  for (int ni = 0; ni < 4; ++ni) {
    const int n = n0 + ni * 16 + (lane & 15);
#pragma unroll
    for (int j = 0; j < 8; ++j) {
      int r  = rbase + j;
      int bb = r >> 9;                             // / S
      int ss = r & (RNN_S - 1);                    // % S
      xw[((size_t)ss * RNN_B + bb) * RNN_H + n] = (bf16_t)acc[ni][j];
    }
  }
}

// ---- phase 2: one recurrence step  h_out = tanh(xw_t + h_in U^T + bias) ----
// 32x32 block per wave: 2 A frags x 2 B frags -> 4 WMMAs per K-step.
// 64 waves/step (2 M-groups x 32 N-groups); U_h stays L2-resident.
__global__ void __launch_bounds__(256)
k_step(const bf16_t* __restrict__ h_in,   // [B, H] bf16
       bf16_t* __restrict__ h_out,        // [B, H] bf16
       const bf16_t* __restrict__ Uhb,    // [H, H] bf16
       const bf16_t* __restrict__ xw_t,   // [B, H] bf16 slab for this t
       const float* __restrict__ b_u,
       const float* __restrict__ b_h) {
  const int lane = threadIdx.x & 31;
  const int wave = threadIdx.x >> 5;
  const int w  = blockIdx.x * 8 + wave;            // 0..63
  const int ng = w & 31, mg = w >> 5;              // 32 N-groups, 2 M-groups
  const int m0 = mg * 32, n0 = ng * 32;

  v8f acc[2][2];
#pragma unroll
  for (int mi = 0; mi < 2; ++mi)
#pragma unroll
    for (int ni = 0; ni < 2; ++ni) {
      const int rb = m0 + mi * 16 + ((lane >> 4) << 3);
      const int n  = n0 + ni * 16 + (lane & 15);
      const float bias = b_u[n] + b_h[n];
#pragma unroll
      for (int j = 0; j < 8; ++j)
        acc[mi][ni][j] = bias + (float)xw_t[(size_t)(rb + j) * RNN_H + n];
    }

#pragma unroll 2
  for (int k0 = 0; k0 < RNN_H; k0 += 32) {
    v16bf a0 = load_a_bf16(h_in, RNN_H, m0,      k0, lane);
    v16bf a1 = load_a_bf16(h_in, RNN_H, m0 + 16, k0, lane);
    v16bf b0 = load_b_w(Uhb, RNN_H, n0,      k0, lane);
    v16bf b1 = load_b_w(Uhb, RNN_H, n0 + 16, k0, lane);
    acc[0][0] = wmma_bf16(a0, b0, acc[0][0]);
    acc[0][1] = wmma_bf16(a0, b1, acc[0][1]);
    acc[1][0] = wmma_bf16(a1, b0, acc[1][0]);
    acc[1][1] = wmma_bf16(a1, b1, acc[1][1]);
  }

#pragma unroll
  for (int mi = 0; mi < 2; ++mi)
#pragma unroll
    for (int ni = 0; ni < 2; ++ni) {
      const int rb = m0 + mi * 16 + ((lane >> 4) << 3);
      const int n  = n0 + ni * 16 + (lane & 15);
#pragma unroll
      for (int j = 0; j < 8; ++j)
        h_out[(size_t)(rb + j) * RNN_H + n] = (bf16_t)fast_tanh(acc[mi][ni][j]);
    }
}

// ---- phase 3: y = h_last @ Wy^T + b_wy + b_y  (f32 out) --------------------
// 32x32 per wave: 32 waves (2 M-groups x 16 N-groups) -> 4 blocks.
__global__ void __launch_bounds__(256)
k_out(const bf16_t* __restrict__ h,       // [B, H] bf16
      const bf16_t* __restrict__ Wyb,     // [O, H] bf16
      const float* __restrict__ b_wy, const float* __restrict__ b_y,
      float* __restrict__ y) {            // [B, O] f32
  const int lane = threadIdx.x & 31;
  const int wave = threadIdx.x >> 5;
  const int w  = blockIdx.x * 8 + wave;            // 0..31
  const int ng = w & 15, mg = w >> 4;              // 16 N-groups, 2 M-groups
  const int m0 = mg * 32, n0 = ng * 32;

  v8f acc[2][2];
#pragma unroll
  for (int mi = 0; mi < 2; ++mi)
#pragma unroll
    for (int ni = 0; ni < 2; ++ni) {
      const int n = n0 + ni * 16 + (lane & 15);
      const float bias = b_wy[n] + b_y[n];
#pragma unroll
      for (int j = 0; j < 8; ++j) acc[mi][ni][j] = bias;
    }

#pragma unroll 2
  for (int k0 = 0; k0 < RNN_H; k0 += 32) {
    v16bf a0 = load_a_bf16(h, RNN_H, m0,      k0, lane);
    v16bf a1 = load_a_bf16(h, RNN_H, m0 + 16, k0, lane);
    v16bf b0 = load_b_w(Wyb, RNN_H, n0,      k0, lane);
    v16bf b1 = load_b_w(Wyb, RNN_H, n0 + 16, k0, lane);
    acc[0][0] = wmma_bf16(a0, b0, acc[0][0]);
    acc[0][1] = wmma_bf16(a0, b1, acc[0][1]);
    acc[1][0] = wmma_bf16(a1, b0, acc[1][0]);
    acc[1][1] = wmma_bf16(a1, b1, acc[1][1]);
  }

#pragma unroll
  for (int mi = 0; mi < 2; ++mi)
#pragma unroll
    for (int ni = 0; ni < 2; ++ni) {
      const int rb = m0 + mi * 16 + ((lane >> 4) << 3);
      const int n  = n0 + ni * 16 + (lane & 15);
#pragma unroll
      for (int j = 0; j < 8; ++j)
        y[(size_t)(rb + j) * RNN_O + n] = acc[mi][ni][j];
    }
}

// ---------------------------------------------------------------------------
extern "C" void kernel_launch(void* const* d_in, const int* in_sizes, int n_in,
                              void* d_out, int out_size, void* d_ws, size_t ws_size,
                              hipStream_t stream) {
  (void)in_sizes; (void)n_in; (void)out_size; (void)ws_size;
  const float* x    = (const float*)d_in[0];   // [B,S,I]
  const float* W_h  = (const float*)d_in[1];   // [H,I]
  const float* U_h  = (const float*)d_in[2];   // [H,H]
  const float* b_u  = (const float*)d_in[3];   // [H]
  const float* b_h  = (const float*)d_in[4];   // [H]
  const float* W_y  = (const float*)d_in[5];   // [O,H]
  const float* b_wy = (const float*)d_in[6];   // [O]
  const float* b_y  = (const float*)d_in[7];   // [O]

  // Workspace layout (bytes):
  //   [0, 1MB)            Wh bf16   (1024*512*2)
  //   [1MB, 3MB)          Uh bf16   (1024*1024*2)
  //   [3MB, 4MB)          Wy bf16   (512*1024*2)
  //   [4MB, 4MB+256KB)    h ping-pong buffers, 2 x [64,1024] bf16
  //   [4MB+256KB, +64MB)  xW bf16   [S][B][H]
  char* ws = (char*)d_ws;
  bf16_t* Whb  = (bf16_t*)(ws);
  bf16_t* Uhb  = (bf16_t*)(ws + (size_t)(1 << 20));
  bf16_t* Wyb  = (bf16_t*)(ws + (size_t)(3 << 20));
  bf16_t* hbuf = (bf16_t*)(ws + (size_t)(4 << 20));
  bf16_t* xw   = (bf16_t*)(ws + (size_t)(4 << 20) + (size_t)(256 << 10));
  bf16_t* h0 = hbuf;                   // buffer 0
  bf16_t* h1 = hbuf + RNN_B * RNN_H;   // buffer 1

  // 1) Convert weights to bf16, zero h0.
  {
    const int total = RNN_H * RNN_I + RNN_H * RNN_H + RNN_O * RNN_H + RNN_B * RNN_H;
    k_prep<<<(total + 255) / 256, 256, 0, stream>>>(W_h, U_h, W_y, Whb, Uhb, Wyb, h0);
  }

  // 2) Input projection: 2048 M-tiles x 16 N-groups = 32768 waves, 8/block.
  k_xw<<<32768 / 8, 256, 0, stream>>>(x, Whb, xw);

  // 3) Sequential recurrence: 512 step kernels (graph capture amortizes launch).
  for (int t = 0; t < RNN_S; ++t) {
    const bf16_t* hin  = (t & 1) ? h1 : h0;
    bf16_t*       hout = (t & 1) ? h0 : h1;
    k_step<<<8, 256, 0, stream>>>(hin, hout, Uhb,
                                  xw + (size_t)t * RNN_B * RNN_H, b_u, b_h);
  }
  // S=512 even -> last write (t=511, odd) landed in buffer 0.

  // 4) Output projection with fused biases.
  k_out<<<4, 256, 0, stream>>>(h0, Wyb, b_wy, b_y, (float*)d_out);
}
